// LocalFeatureFusion_29025388986479
// MI455X (gfx1250) — matrix operand order repaired and checked by reading, not compiled, not run on hardware
//
#include <hip/hip_runtime.h>
#include <cstdint>
#include <cstddef>

// Problem constants (match reference)
#define BB 2
#define NQ 8192
#define LL 8192
#define CC 128
#define KK 8
#define FF 512
#define R2C 0.16f   // RADIUS^2

typedef _Float16 half_t;
typedef _Float16 v16h __attribute__((ext_vector_type(16)));
typedef _Float16 h8   __attribute__((ext_vector_type(8)));
typedef float    v8f  __attribute__((ext_vector_type(8)));
typedef float    f4   __attribute__((ext_vector_type(4)));

// Detect CDNA5 async-to-LDS builtins (gfx1250)
#if defined(__has_builtin)
#if __has_builtin(__builtin_amdgcn_global_load_async_to_lds_b32) && \
    __has_builtin(__builtin_amdgcn_s_wait_asynccnt)
#define USE_ASYNC_LDS 1
#endif
#endif
#ifndef USE_ASYNC_LDS
#define USE_ASYNC_LDS 0
#endif

typedef __attribute__((address_space(1))) int glob_i32;
typedef __attribute__((address_space(3))) int lds_i32;

// ---------------------------------------------------------------------------
// fp32 -> fp16 elementwise conversion (only for things not B-packed)
// ---------------------------------------------------------------------------
__global__ void conv_f16_kernel(const float* __restrict__ s, half_t* __restrict__ d, int n) {
  int i = blockIdx.x * blockDim.x + threadIdx.x;
  if (i < n) d[i] = (half_t)s[i];
}

// ---------------------------------------------------------------------------
// Pack a [Kd,N] fp32 weight into f16 "B-tile lane-major" layout:
// tile tg = s*(N/16)+t covers k in [32s,32s+32), n in [16t,16t+16).
// Lane l of the consuming wave reads 16 contiguous halves:
//   Wp[(tg*32 + l)*16 + j] = W[(32s + 8*(l>>4) + (j<8 ? j : j+8))*N + (16t + (l&15))]
// so each B tile is two global_load_b128 per lane instead of 16 u16 loads.
// ---------------------------------------------------------------------------
__global__ void pack_b_kernel(const float* __restrict__ W, half_t* __restrict__ Wp,
                              int Kd, int N) {
  int oi = blockIdx.x * blockDim.x + threadIdx.x;
  if (oi >= Kd * N) return;
  int j    = oi & 15;
  int l    = (oi >> 4) & 31;
  int tile = oi >> 9;
  int nt   = N >> 4;
  int t = tile % nt;
  int s = tile / nt;
  int k = s * 32 + (l >> 4) * 8 + (j < 8 ? j : j + 8);
  int n = t * 16 + (l & 15);
  Wp[oi] = (half_t)W[(size_t)k * N + n];
}

// ---------------------------------------------------------------------------
// A-tile loaders (documented 16x32 f16 A layout; per lane: two contiguous
// 8-element runs at [0..7] and [16..23] relative to s*32 + hi*8)
// ---------------------------------------------------------------------------
__device__ __forceinline__ v16h load_a_tile(const float* xr) {
  f4 c0 = *(const f4*)xr;
  f4 c1 = *(const f4*)(xr + 4);
  f4 c2 = *(const f4*)(xr + 16);
  f4 c3 = *(const f4*)(xr + 20);
  v16h a;
#pragma unroll
  for (int j = 0; j < 4; ++j) {
    a[j]      = (half_t)c0[j];
    a[4 + j]  = (half_t)c1[j];
    a[8 + j]  = (half_t)c2[j];
    a[12 + j] = (half_t)c3[j];
  }
  return a;
}
__device__ __forceinline__ v16h load_a_tile(const half_t* xr) {
  h8 h0 = *(const h8*)xr;
  h8 h1 = *(const h8*)(xr + 16);
  v16h a;
#pragma unroll
  for (int j = 0; j < 8; ++j) { a[j] = h0[j]; a[8 + j] = h1[j]; }
  return a;
}

// ---------------------------------------------------------------------------
// Radius-kNN: one thread per query, kv_xyz tiled through LDS (async on CDNA5),
// branchless 8-deep insertion sort (ascending d2, stable on ties).
// ---------------------------------------------------------------------------
#define TL 512
__global__ void knn_kernel(const float* __restrict__ q_xyz, const float* __restrict__ kv_xyz,
                           const unsigned char* __restrict__ kv_pad, int* __restrict__ idx_out) {
  __shared__ float sxyz[TL * 3];
  __shared__ unsigned char sp[TL];
  int row = blockIdx.x * blockDim.x + threadIdx.x;   // global query row in [0, B*N)
  int b = row / NQ;
  float qx = q_xyz[row * 3 + 0];
  float qy = q_xyz[row * 3 + 1];
  float qz = q_xyz[row * 3 + 2];
  float bd[KK]; int bi[KK];
#pragma unroll
  for (int j = 0; j < KK; ++j) { bd[j] = 3.0e38f; bi[j] = -1; }

  for (int t0 = 0; t0 < LL; t0 += TL) {
    __syncthreads();
#if USE_ASYNC_LDS
    {
      const float* gsrc = kv_xyz + (size_t)(b * LL + t0) * 3;
      for (int i = threadIdx.x; i < TL * 3; i += blockDim.x) {
        __builtin_amdgcn_global_load_async_to_lds_b32(
            (glob_i32*)(gsrc + i), (lds_i32*)&sxyz[i], 0, 0);
      }
      for (int i = threadIdx.x; i < TL; i += blockDim.x)
        sp[i] = kv_pad[b * LL + t0 + i];
      __builtin_amdgcn_s_wait_asynccnt(0);
    }
#else
    for (int i = threadIdx.x; i < TL * 3; i += blockDim.x)
      sxyz[i] = kv_xyz[(size_t)(b * LL + t0) * 3 + i];
    for (int i = threadIdx.x; i < TL; i += blockDim.x)
      sp[i] = kv_pad[b * LL + t0 + i];
#endif
    __syncthreads();
    for (int i = 0; i < TL; ++i) {
      float dx = qx - sxyz[3 * i + 0];
      float dy = qy - sxyz[3 * i + 1];
      float dz = qz - sxyz[3 * i + 2];
      float d2 = dx * dx + dy * dy + dz * dz;
      if (d2 <= R2C && !sp[i] && d2 < bd[KK - 1]) {
        float dv = d2; int iv = t0 + i;
#pragma unroll
        for (int j = 0; j < KK; ++j) {
          bool sw = dv < bd[j];
          float td = bd[j]; int ti = bi[j];
          bd[j] = sw ? dv : td; bi[j] = sw ? iv : ti;
          dv    = sw ? td : dv; iv    = sw ? ti : iv;
        }
      }
    }
  }
#pragma unroll
  for (int j = 0; j < KK; ++j)
    idx_out[row * KK + j] = (bd[j] <= R2C) ? bi[j] : -1;
}

// ---------------------------------------------------------------------------
// Generic WMMA GEMM:  Y[M,N] = act(X[M,Kd] @ W[Kd,N] + bias) (+ res)
// W is B-packed (pack_b_kernel). Wave computes a 16x64 output tile.
// Block = 4 waves = 64 rows. ACT: 0=none, 1=relu, 2=exact gelu.
// ---------------------------------------------------------------------------
template <typename XT, typename YT, int ACT, bool RES>
__global__ void gemm_wmma_kernel(const XT* __restrict__ X, const half_t* __restrict__ Wp,
                                 const float* __restrict__ bias, const float* __restrict__ res,
                                 YT* __restrict__ Y, int M, int Kd, int N) {
  int lane = threadIdx.x & 31;
  int wav  = threadIdx.x >> 5;
  int row0 = (blockIdx.x * 4 + wav) * 16;
  int col0 = blockIdx.y * 64;
  int m  = lane & 15;
  int hi = lane >> 4;
  int nt = N >> 4;

  v8f acc[4];
#pragma unroll
  for (int tt = 0; tt < 4; ++tt) {
    float bv = bias ? bias[col0 + 16 * tt + m] : 0.0f;
#pragma unroll
    for (int v = 0; v < 8; ++v) acc[tt][v] = bv;
  }

  for (int s = 0; s < Kd / 32; ++s) {
    v16h a = load_a_tile(X + (size_t)(row0 + m) * Kd + s * 32 + hi * 8);
#pragma unroll
    for (int tt = 0; tt < 4; ++tt) {
      int tg = s * nt + (col0 >> 4) + tt;
      v16h bm = *(const v16h*)(Wp + ((size_t)tg * 32 + lane) * 16);
      acc[tt] = __builtin_amdgcn_wmma_f32_16x16x32_f16(
          false, a, false, bm, (short)0, acc[tt], false, false);
    }
  }

#pragma unroll
  for (int tt = 0; tt < 4; ++tt) {
    int n = col0 + 16 * tt + m;
#pragma unroll
    for (int v = 0; v < 8; ++v) {
      int r = row0 + v + 8 * hi;
      float val = acc[tt][v];
      if (ACT == 1) val = fmaxf(val, 0.0f);
      if (ACT == 2) val = 0.5f * val * (1.0f + erff(val * 0.70710678118654752f));
      if (RES) val += res[(size_t)r * N + n];
      Y[(size_t)r * N + n] = (YT)val;
    }
  }
}

// ---------------------------------------------------------------------------
// Fused per-edge vector attention. One wave = 2 queries (16 edge rows).
// delta = relu(relu(rel@pw1+pb1)@pw2+pb2)        (WMMA, D layout -> LDS)
// alpha = relu((a_dst - a_src + delta)@attn_w+b) (WMMA)
// per-channel softmax over the 8 edges is a per-lane reduction (D layout!)
// pre1 = q_feat + sum_k attn * (x_j + delta)
// ---------------------------------------------------------------------------
__global__ void edge_attn_kernel(const float* __restrict__ q_xyz, const float* __restrict__ q_feat,
                                 const float* __restrict__ kv_xyz, const int* __restrict__ idx,
                                 const float* __restrict__ kv_src, const float* __restrict__ kv_lin,
                                 const float* __restrict__ a_dst,
                                 const float* __restrict__ pos_w1, const float* __restrict__ pos_b1,
                                 const half_t* __restrict__ pos_w2p, const float* __restrict__ pos_b2,
                                 const half_t* __restrict__ attn_wp, const float* __restrict__ attn_b,
                                 float* __restrict__ pre1) {
  __shared__ __attribute__((aligned(16))) float sDelta[4][16][CC];  // per-wave 16x128 delta
  int lane = threadIdx.x & 31;
  int wav  = threadIdx.x >> 5;
  int qg = blockIdx.x * 8 + wav * 2;    // first query row of this wave's pair
  int b  = qg / NQ;
  int m  = lane & 15;                   // edge-row for A tiles (0..15)
  int hi = lane >> 4;

  // this lane's edge (row m): query qg + m/8, neighbor slot m%8
  int e_q = qg + (m >> 3);
  int e_k = m & 7;
  int id_raw = idx[e_q * KK + e_k];
  bool valid = id_raw >= 0;
  int ide = valid ? id_raw : 0;
  unsigned vmask = (unsigned)__ballot(valid);   // bit e = validity of edge e (0..15)

  float rx = q_xyz[e_q * 3 + 0] - kv_xyz[(size_t)(b * LL + ide) * 3 + 0];
  float ry = q_xyz[e_q * 3 + 1] - kv_xyz[(size_t)(b * LL + ide) * 3 + 1];
  float rz = q_xyz[e_q * 3 + 2] - kv_xyz[(size_t)(b * LL + ide) * 3 + 2];

  // hidden = relu(rel @ pos_w1 + pos_b1), built directly in A layout (f16)
  v16h aH[4];
#pragma unroll
  for (int s = 0; s < 4; ++s) {
#pragma unroll
    for (int v = 0; v < 8; ++v) {
      int kb = s * 32 + ((v < 4) ? 2 * v : 8 + 2 * v) + hi * 8;
#pragma unroll
      for (int p = 0; p < 2; ++p) {
        int k = kb + p;
        float h = rx * pos_w1[k] + ry * pos_w1[CC + k] + rz * pos_w1[2 * CC + k] + pos_b1[k];
        aH[s][2 * v + p] = (half_t)fmaxf(h, 0.0f);
      }
    }
  }

  // delta tiles -> LDS (D layout: lane holds col n, rows v + 8*hi)
#pragma unroll
  for (int t = 0; t < 8; ++t) {
    int n = 16 * t + m;
    v8f acc;
    float bv = pos_b2[n];
#pragma unroll
    for (int v = 0; v < 8; ++v) acc[v] = bv;
#pragma unroll
    for (int s = 0; s < 4; ++s) {
      v16h bm = *(const v16h*)(pos_w2p + ((size_t)(s * 8 + t) * 32 + lane) * 16);
      acc = __builtin_amdgcn_wmma_f32_16x16x32_f16(
          false, aH[s], false, bm, (short)0, acc, false, false);
    }
#pragma unroll
    for (int v = 0; v < 8; ++v)
      sDelta[wav][v + 8 * hi][n] = fmaxf(acc[v], 0.0f);
  }
  __syncthreads();

  // A_alpha = a_dst - a_src + delta  (A layout, row m) -- vectorized chunks
  v16h aA[4];
  {
    const float* ad = a_dst + (size_t)e_q * CC;
    const float* as = kv_src + (size_t)(b * LL + ide) * CC;
    const float* sd = &sDelta[wav][m][0];
#pragma unroll
    for (int s = 0; s < 4; ++s) {
      int base = s * 32 + hi * 8;
      f4 A0 = *(const f4*)(ad + base),      S0 = *(const f4*)(as + base),      D0 = *(const f4*)(sd + base);
      f4 A1 = *(const f4*)(ad + base + 4),  S1 = *(const f4*)(as + base + 4),  D1 = *(const f4*)(sd + base + 4);
      f4 A2 = *(const f4*)(ad + base + 16), S2 = *(const f4*)(as + base + 16), D2 = *(const f4*)(sd + base + 16);
      f4 A3 = *(const f4*)(ad + base + 20), S3 = *(const f4*)(as + base + 20), D3 = *(const f4*)(sd + base + 20);
#pragma unroll
      for (int j = 0; j < 4; ++j) {
        aA[s][j]      = (half_t)(A0[j] - S0[j] + D0[j]);
        aA[s][4 + j]  = (half_t)(A1[j] - S1[j] + D1[j]);
        aA[s][8 + j]  = (half_t)(A2[j] - S2[j] + D2[j]);
        aA[s][12 + j] = (half_t)(A3[j] - S3[j] + D3[j]);
      }
    }
  }

  // neighbor ids for the 8 accumulator rows of this lane's query
  int idv[8];
#pragma unroll
  for (int v = 0; v < 8; ++v) idv[v] = __shfl(ide, v + 8 * hi, 32);

  int outrow = qg + hi;
  const float* qf = q_feat + (size_t)outrow * CC;
  float* pr = pre1 + (size_t)outrow * CC;

#pragma unroll
  for (int t = 0; t < 8; ++t) {
    int n = 16 * t + m;
    v8f acc;
    float bv = attn_b[n];
#pragma unroll
    for (int v = 0; v < 8; ++v) acc[v] = bv;
#pragma unroll
    for (int s = 0; s < 4; ++s) {
      v16h bm = *(const v16h*)(attn_wp + ((size_t)(s * 8 + t) * 32 + lane) * 16);
      acc = __builtin_amdgcn_wmma_f32_16x16x32_f16(
          false, aA[s], false, bm, (short)0, acc, false, false);
    }
    // per-lane softmax over the 8 edges (channel n of query outrow)
    float sc[8]; float mx = -3.0e38f;
#pragma unroll
    for (int v = 0; v < 8; ++v) {
      bool ev = (vmask >> (v + 8 * hi)) & 1u;
      sc[v] = ev ? fmaxf(acc[v], 0.0f) : -3.0e38f;
      mx = fmaxf(mx, sc[v]);
    }
    float mxe = (mx > -1.0e38f) ? mx : 0.0f;
    float ex[8]; float den = 0.0f;
#pragma unroll
    for (int v = 0; v < 8; ++v) {
      ex[v] = (sc[v] > -1.0e38f) ? expf(sc[v] - mxe) : 0.0f;
      den += ex[v];
    }
    den = fmaxf(den, 1e-12f);
    float agg = 0.0f;
#pragma unroll
    for (int v = 0; v < 8; ++v) {
      float xj = kv_lin[(size_t)(b * LL + idv[v]) * CC + n];
      float dl = sDelta[wav][v + 8 * hi][n];
      agg += (ex[v] / den) * (xj + dl);
    }
    pr[n] = qf[n] + agg;
  }
}

// ---------------------------------------------------------------------------
// LayerNorm over C=128: one wave per row, contiguous float4 per lane
// ---------------------------------------------------------------------------
__global__ void ln_kernel(const float* __restrict__ x, const float* __restrict__ g,
                          const float* __restrict__ bt, float* __restrict__ y, int Mrows) {
  int row = blockIdx.x * (blockDim.x >> 5) + (threadIdx.x >> 5);
  int lane = threadIdx.x & 31;
  if (row >= Mrows) return;
  const float* xr = x + (size_t)row * CC;
  f4 v  = *(const f4*)(xr + lane * 4);
  f4 gg = *(const f4*)(g + lane * 4);
  f4 bb = *(const f4*)(bt + lane * 4);
  float s = v[0] + v[1] + v[2] + v[3];
  float sq = v[0]*v[0] + v[1]*v[1] + v[2]*v[2] + v[3]*v[3];
#pragma unroll
  for (int o = 16; o > 0; o >>= 1) {
    s  += __shfl_xor(s, o, 32);
    sq += __shfl_xor(sq, o, 32);
  }
  float mean = s * (1.0f / CC);
  float var  = sq * (1.0f / CC) - mean * mean;
  float inv  = rsqrtf(var + 1e-5f);
  f4 out;
#pragma unroll
  for (int i = 0; i < 4; ++i) out[i] = (v[i] - mean) * inv * gg[i] + bb[i];
  *(f4*)(y + (size_t)row * CC + lane * 4) = out;
}

// ---------------------------------------------------------------------------
// Host driver
// ---------------------------------------------------------------------------
extern "C" void kernel_launch(void* const* d_in, const int* in_sizes, int n_in,
                              void* d_out, int out_size, void* d_ws, size_t ws_size,
                              hipStream_t stream) {
  (void)in_sizes; (void)n_in; (void)out_size; (void)ws_size;
  const float* q_xyz   = (const float*)d_in[0];
  const float* q_feat  = (const float*)d_in[1];
  const float* kv_xyz  = (const float*)d_in[2];
  const float* kv_feat = (const float*)d_in[3];
  const unsigned char* kv_pad = (const unsigned char*)d_in[4];
  const float* pos_w1 = (const float*)d_in[5];
  const float* pos_b1 = (const float*)d_in[6];
  const float* pos_w2 = (const float*)d_in[7];
  const float* pos_b2 = (const float*)d_in[8];
  const float* attn_w = (const float*)d_in[9];
  const float* attn_b = (const float*)d_in[10];
  const float* lin_w     = (const float*)d_in[11];
  const float* lin_src_w = (const float*)d_in[12];
  const float* lin_dst_w = (const float*)d_in[13];
  const float* ln1_g = (const float*)d_in[14];
  const float* ln1_b = (const float*)d_in[15];
  const float* ffn_w1 = (const float*)d_in[16];
  const float* ffn_b1 = (const float*)d_in[17];
  const float* ffn_w2 = (const float*)d_in[18];
  const float* ffn_b2 = (const float*)d_in[19];
  const float* ln2_g = (const float*)d_in[20];
  const float* ln2_b = (const float*)d_in[21];

  char* ws = (char*)d_ws;
  size_t off = 0;
  auto take = [&](size_t bytes) -> char* {
    char* p = ws + off;
    off = (off + bytes + 255) & ~(size_t)255;
    return p;
  };
  half_t* w_pos2 = (half_t*)take((size_t)CC * CC * 2);   // B-packed
  half_t* w_attn = (half_t*)take((size_t)CC * CC * 2);   // B-packed
  half_t* w_lin  = (half_t*)take((size_t)CC * CC * 2);   // B-packed
  half_t* w_src  = (half_t*)take((size_t)CC * CC * 2);   // B-packed
  half_t* w_dst  = (half_t*)take((size_t)CC * CC * 2);   // B-packed
  half_t* w_f1   = (half_t*)take((size_t)CC * FF * 2);   // B-packed
  half_t* w_f2   = (half_t*)take((size_t)FF * CC * 2);   // B-packed
  int*   idxb   = (int*)take((size_t)BB * NQ * KK * 4);
  float* kv_src = (float*)take((size_t)BB * LL * CC * 4);
  float* kv_lin = (float*)take((size_t)BB * LL * CC * 4);
  float* a_dst  = (float*)take((size_t)BB * NQ * CC * 4);
  float* buf1   = (float*)take((size_t)BB * NQ * CC * 4);
  // aliases for buffers that are dead by the time these are live:
  half_t* h1   = (half_t*)kv_src;  // [M,512] f16 = 16.8MB over kv_src+kv_lin
  float*  pre2 = a_dst;            // [M,128] f32

  const int MR = BB * NQ;  // 16384 rows (== B*L too)

  // weight packing to f16 B-tile layout
  pack_b_kernel<<<(CC * CC + 255) / 256, 256, 0, stream>>>(pos_w2, w_pos2, CC, CC);
  pack_b_kernel<<<(CC * CC + 255) / 256, 256, 0, stream>>>(attn_w, w_attn, CC, CC);
  pack_b_kernel<<<(CC * CC + 255) / 256, 256, 0, stream>>>(lin_w, w_lin, CC, CC);
  pack_b_kernel<<<(CC * CC + 255) / 256, 256, 0, stream>>>(lin_src_w, w_src, CC, CC);
  pack_b_kernel<<<(CC * CC + 255) / 256, 256, 0, stream>>>(lin_dst_w, w_dst, CC, CC);
  pack_b_kernel<<<(CC * FF + 255) / 256, 256, 0, stream>>>(ffn_w1, w_f1, CC, FF);
  pack_b_kernel<<<(FF * CC + 255) / 256, 256, 0, stream>>>(ffn_w2, w_f2, FF, CC);

  // radius-kNN
  knn_kernel<<<MR / 256, 256, 0, stream>>>(q_xyz, kv_xyz, kv_pad, idxb);

  // projections (WMMA)
  dim3 g128(MR / 64, CC / 64);
  gemm_wmma_kernel<float, float, 0, false><<<g128, 128, 0, stream>>>(
      kv_feat, w_src, nullptr, nullptr, kv_src, MR, CC, CC);
  gemm_wmma_kernel<float, float, 0, false><<<g128, 128, 0, stream>>>(
      kv_feat, w_lin, nullptr, nullptr, kv_lin, MR, CC, CC);
  gemm_wmma_kernel<float, float, 0, false><<<g128, 128, 0, stream>>>(
      q_feat, w_dst, nullptr, nullptr, a_dst, MR, CC, CC);

  // fused edge attention -> pre1 (= q_feat + agg), stored in buf1
  edge_attn_kernel<<<MR / 8, 128, 0, stream>>>(
      q_xyz, q_feat, kv_xyz, idxb, kv_src, kv_lin, a_dst,
      pos_w1, pos_b1, w_pos2, pos_b2, w_attn, attn_b, buf1);

  // LN1 (in place): buf1 = out1
  ln_kernel<<<MR / 8, 256, 0, stream>>>(buf1, ln1_g, ln1_b, buf1, MR);

  // FFN1: h1 = gelu(out1 @ ffn_w1 + b1)   [M,512] f16
  dim3 gffn1(MR / 64, FF / 64);
  gemm_wmma_kernel<float, half_t, 2, false><<<gffn1, 128, 0, stream>>>(
      buf1, w_f1, ffn_b1, nullptr, h1, MR, CC, FF);

  // FFN2: pre2 = out1 + h1 @ ffn_w2 + b2
  gemm_wmma_kernel<half_t, float, 0, true><<<g128, 128, 0, stream>>>(
      h1, w_f2, ffn_b2, buf1, pre2, MR, FF, CC);

  // LN2 -> final output
  ln_kernel<<<MR / 8, 256, 0, stream>>>(pre2, ln2_g, ln2_b, (float*)d_out, MR);
}
